// SpaceTimeTransformer_79989470921243
// MI455X (gfx1250) — matrix operand order repaired and checked by reading, not compiled, not run on hardware
//
#include <hip/hip_runtime.h>

// ---------------------------------------------------------------------------
// Problem constants (from reference)
// ---------------------------------------------------------------------------
#define DMODEL 256
#define NHEADS 8
#define DKH    32
#define DFF    1024
#define NB_    2
#define SS_    144
#define CC_    8
#define TT_    24
#define NTOK   (NB_*SS_*CC_*TT_)   // 55296 tokens
#define LN_EPS 1e-6f

typedef __attribute__((ext_vector_type(16))) _Float16 v16h;
typedef __attribute__((ext_vector_type(8)))  float    v8f;

union AFrag { v16h v; _Float16 h[16]; unsigned u[8]; uint4 q[2]; };

__device__ __forceinline__ v8f wmma16(const AFrag& a, const AFrag& b, v8f c) {
  // (neg_a, A, neg_b, B, c_mod, C, reuse_a, reuse_b)
  return __builtin_amdgcn_wmma_f32_16x16x32_f16(false, a.v, false, b.v,
                                                (short)0, c, false, false);
}

// A-operand k-swizzle within a 32-chunk: [0..7 | 16..23 | 8..15 | 24..31].
// Lane<16 then reads halves [0,16) contiguously (= k {0..7,16..23}),
// lane>=16 reads [16,32) (= k {8..15,24..31}), matching ISA 16-bit A layout.
__device__ __forceinline__ int aswz(int col) {          // col in [0,32)
  return ((col & 8) << 1) | ((col & 16) >> 1) | (col & 7);
}

// ---------------------------------------------------------------------------
// fp32 -> f16 weight conversion WITH transpose: dst[(m,n,k)] = src[(m,k,n)]
// dst holds nmat matrices of shape N x K (k-contiguous rows = WMMA B-friendly)
// ---------------------------------------------------------------------------
__global__ void k_cvt_t(const float* __restrict__ src, _Float16* __restrict__ dst,
                        int K, int N, int nmat) {
  int i = blockIdx.x * 256 + threadIdx.x;
  int tot = K * N;
  if (i < tot * nmat) {
    int m = i / tot, rem = i - m * tot;
    int n = rem / K, k = rem - n * K;
    dst[i] = (_Float16)src[m * tot + k * N + n];
  }
}

// ---------------------------------------------------------------------------
// LayerNorm: one wave32 per token (256 features -> 8 per lane), f16 out
// ---------------------------------------------------------------------------
__global__ __launch_bounds__(256) void k_ln(const float* __restrict__ x,
                                            const float* __restrict__ g,
                                            const float* __restrict__ bta,
                                            _Float16* __restrict__ y) {
  int lane = threadIdx.x & 31, wid = threadIdx.x >> 5;
  int tok  = blockIdx.x * 8 + wid;
  const float* xr = x + (size_t)tok * DMODEL + lane * 8;
  float v[8];
  float s = 0.f;
#pragma unroll
  for (int i = 0; i < 8; ++i) { v[i] = xr[i]; s += v[i]; }
#pragma unroll
  for (int o = 16; o > 0; o >>= 1) s += __shfl_xor(s, o, 32);
  float mean = s * (1.0f / DMODEL);
  float q = 0.f;
#pragma unroll
  for (int i = 0; i < 8; ++i) { float d = v[i] - mean; q += d * d; }
#pragma unroll
  for (int o = 16; o > 0; o >>= 1) q += __shfl_xor(q, o, 32);
  float rstd = rsqrtf(q * (1.0f / DMODEL) + LN_EPS);
  _Float16* yr = y + (size_t)tok * DMODEL + lane * 8;
  const float* gr = g + lane * 8;
  const float* br = bta + lane * 8;
#pragma unroll
  for (int i = 0; i < 8; ++i)
    yr[i] = (_Float16)(gr[i] * (v[i] - mean) * rstd + br[i]);
}

// ---------------------------------------------------------------------------
// WMMA GEMM: C[M,N] = A[M,K](f16) * Bt[N,K](f16, pre-transposed) + bias
//   MODE 0: out f16 = acc + bias            (QKV projections)
//   MODE 1: out f16 = relu(acc + bias)      (FFN1)
//   MODE 2: out f32 = resid + acc + bias    (out-proj / FFN2, residual add)
// Block = 256 threads (8 waves, 4x2 wave grid), tile 128x64, KC=32.
// Each wave: 32x32 slab = 2 A-frags x 2 B-frags -> 4 WMMAs per K-step
// (2 ds_load_b128 per WMMA; register-level A/B reuse).
// M%128==0, N%64==0, K%32==0.
// ---------------------------------------------------------------------------
template <int MODE>
__global__ __launch_bounds__(256) void k_gemm(const _Float16* __restrict__ A,
                                              const _Float16* __restrict__ Bt,
                                              const float* __restrict__ bias,
                                              void* __restrict__ outp,
                                              const float* __restrict__ resid,
                                              int M, int N, int K) {
  __shared__ __align__(16) _Float16 Al[128][32];  // A tile, k-chunk swizzled
  __shared__ __align__(16) _Float16 Bl[64][32];   // Bt tile (rows = n, k-chunk swizzled)
  const int tid = threadIdx.x, lane = tid & 31, wid = tid >> 5;
  const int m0 = blockIdx.x * 128, n0 = blockIdx.y * 64;
  const int rm = wid & 3;       // 32-row band
  const int cn = wid >> 2;      // 32-col band
  v8f acc00 = {}, acc01 = {}, acc10 = {}, acc11 = {};

  const int tr  = tid >> 2;             // stage row 0..63
  const int tc8 = (tid & 3) << 3;       // stage col chunk {0,8,16,24}
  const int swz8 = (tc8 == 8) ? 16 : (tc8 == 16) ? 8 : tc8;  // group swap
  const int off16 = (lane >= 16) ? 16 : 0;
  const int mrow0 = rm * 32 + (lane & 15);
  const int mrow1 = mrow0 + 16;
  const int ncol  = lane & 15;
  const int brow0 = cn * 32 + ncol;
  const int brow1 = brow0 + 16;

  for (int k0 = 0; k0 < K; k0 += 32) {
    *(uint4*)&Al[tr][swz8]      = *(const uint4*)&A[(size_t)(m0 + tr) * K + k0 + tc8];
    *(uint4*)&Al[tr + 64][swz8] = *(const uint4*)&A[(size_t)(m0 + 64 + tr) * K + k0 + tc8];
    *(uint4*)&Bl[tr][swz8]      = *(const uint4*)&Bt[(size_t)(n0 + tr) * K + k0 + tc8];
    if (k0 + 32 < K) {
      __builtin_prefetch(&A[(size_t)(m0 + tr) * K + k0 + 32 + tc8], 0, 1);
      __builtin_prefetch(&A[(size_t)(m0 + 64 + tr) * K + k0 + 32 + tc8], 0, 1);
      __builtin_prefetch(&Bt[(size_t)(n0 + tr) * K + k0 + 32 + tc8], 0, 1);
    }
    __syncthreads();

    AFrag a0, a1, b0, b1;
    a0.q[0] = *(const uint4*)&Al[mrow0][off16];
    a0.q[1] = *(const uint4*)&Al[mrow0][off16 + 8];
    a1.q[0] = *(const uint4*)&Al[mrow1][off16];
    a1.q[1] = *(const uint4*)&Al[mrow1][off16 + 8];
    b0.q[0] = *(const uint4*)&Bl[brow0][off16];
    b0.q[1] = *(const uint4*)&Bl[brow0][off16 + 8];
    b1.q[0] = *(const uint4*)&Bl[brow1][off16];
    b1.q[1] = *(const uint4*)&Bl[brow1][off16 + 8];

    acc00 = wmma16(a0, b0, acc00);
    acc01 = wmma16(a0, b1, acc01);
    acc10 = wmma16(a1, b0, acc10);
    acc11 = wmma16(a1, b1, acc11);
    __syncthreads();
  }

  const int hi = (lane >= 16) ? 8 : 0;
#pragma unroll
  for (int vv = 0; vv < 8; ++vv) {
    int gm0 = m0 + rm * 32 + vv + hi;
    int gm1 = gm0 + 16;
    int gn0 = n0 + cn * 32 + ncol;
    int gn1 = gn0 + 16;
    float bia0 = bias[gn0], bia1 = bias[gn1];
    float r00 = acc00[vv] + bia0, r01 = acc01[vv] + bia1;
    float r10 = acc10[vv] + bia0, r11 = acc11[vv] + bia1;
    if constexpr (MODE == 1) {
      r00 = fmaxf(r00, 0.f); r01 = fmaxf(r01, 0.f);
      r10 = fmaxf(r10, 0.f); r11 = fmaxf(r11, 0.f);
    }
    if constexpr (MODE <= 1) {
      _Float16* o = (_Float16*)outp;
      o[(size_t)gm0 * N + gn0] = (_Float16)r00;
      o[(size_t)gm0 * N + gn1] = (_Float16)r01;
      o[(size_t)gm1 * N + gn0] = (_Float16)r10;
      o[(size_t)gm1 * N + gn1] = (_Float16)r11;
    } else {
      float* o = (float*)outp;
      size_t i00 = (size_t)gm0 * N + gn0, i01 = (size_t)gm0 * N + gn1;
      size_t i10 = (size_t)gm1 * N + gn0, i11 = (size_t)gm1 * N + gn1;
      o[i00] = resid[i00] + r00;
      o[i01] = resid[i01] + r01;
      o[i10] = resid[i10] + r10;
      o[i11] = resid[i11] + r11;
    }
  }
}

// ---------------------------------------------------------------------------
// Axial multi-head attention. One block per (group, head).
// AXIS 0 = space (L=144), 1 = channel (L=8), 2 = time (L=24).
// LDS: qs (A-swizzled), ks (A-swizzled too: rows already k-contiguous for QK^T),
//      vs_t (transposed V: d-major -> b128 B-fragments for PV),
//      pb (P in A-swizzled order, written by softmax).
// ---------------------------------------------------------------------------
template <int AXIS, int NW>
__global__ __launch_bounds__(NW * 32) void k_attn(const _Float16* __restrict__ Q,
                                                  const _Float16* __restrict__ Km,
                                                  const _Float16* __restrict__ V,
                                                  _Float16* __restrict__ O) {
  constexpr int L   = (AXIS == 0) ? SS_ : (AXIS == 1) ? CC_ : TT_;
  constexpr int STR = (AXIS == 0) ? CC_ * TT_ : (AXIS == 1) ? TT_ : 1;
  constexpr int Lp  = ((L + 15) / 16) * 16;   // k-tile padded
  constexpr int Lpp = ((L + 31) / 32) * 32;   // PV chunk padded
  constexpr int nQT = Lp / 16, nKT = Lp / 16, nKC = Lpp / 32;

  extern __shared__ __align__(16) char smem[];
  _Float16* qs  = (_Float16*)smem;                 // [Lp][32]  (A-swizzled)
  _Float16* ks  = qs + Lp * 32;                    // [Lp][32]  (A-swizzled)
  _Float16* vst = ks + Lp * 32;                    // [32][Lpp] (V transposed)
  float*    sc  = (float*)(vst + 32 * Lpp);        // [NW][16][Lp]
  _Float16* pb  = (_Float16*)(sc + NW * 16 * Lp);  // [NW][16][Lpp] (A-swizzled)

  const int g = blockIdx.x, h = blockIdx.y;
  int base;
  if constexpr (AXIS == 0) {
    int n = g / (CC_ * TT_), rr = g % (CC_ * TT_);
    base = n * (SS_ * CC_ * TT_) + rr;
  } else if constexpr (AXIS == 1) {
    int n = g / (SS_ * TT_), rr = g % (SS_ * TT_);
    int s = rr / TT_, t = rr % TT_;
    base = (n * SS_ + s) * (CC_ * TT_) + t;
  } else {
    base = g * TT_;
  }

  const int tid = threadIdx.x, lane = tid & 31, wid = tid >> 5;
  const int off16 = (lane >= 16) ? 16 : 0;
  const int ncol  = lane & 15;
  const int hi    = (lane >= 16) ? 8 : 0;

  // Cooperative load of Q/K/V head slices into LDS (zero-padded rows)
  for (int idx = tid; idx < Lp * 32; idx += NW * 32) {
    int row = idx >> 5, col = idx & 31;
    _Float16 q = (_Float16)0.f, k = (_Float16)0.f, v = (_Float16)0.f;
    if (row < L) {
      size_t p = (size_t)(base + row * STR) * DMODEL + h * DKH + col;
      q = Q[p]; k = Km[p]; v = V[p];
    }
    qs[row * 32 + aswz(col)] = q;
    ks[row * 32 + aswz(col)] = k;
    vst[col * Lpp + (row & ~31) + aswz(row & 31)] = v;
  }
  __syncthreads();

  const float scale = 0.17677669529663687f;  // 1/sqrt(32)
  constexpr int nIter = (nQT + NW - 1) / NW;

  for (int it = 0; it < nIter; ++it) {
    const int qt = it * NW + wid;
    const bool act = (qt < nQT);

    if (act) {  // ---- scores: Q(16x32) x K^T(32x16) per k-tile
      const int mrow = qt * 16 + (lane & 15);
      AFrag a;
      a.q[0] = *(const uint4*)&qs[mrow * 32 + off16];
      a.q[1] = *(const uint4*)&qs[mrow * 32 + off16 + 8];
      float* scw = sc + wid * 16 * Lp;
      for (int kt = 0; kt < nKT; ++kt) {
        AFrag b;  // B(kk,n) = K[kt*16+n][kk]; K rows are k-contiguous already
        b.q[0] = *(const uint4*)&ks[(kt * 16 + ncol) * 32 + off16];
        b.q[1] = *(const uint4*)&ks[(kt * 16 + ncol) * 32 + off16 + 8];
        v8f c = {};
        c = wmma16(a, b, c);
#pragma unroll
        for (int vv = 0; vv < 8; ++vv)
          scw[(vv + hi) * Lp + kt * 16 + ncol] = c[vv] * scale;
      }
    }
    __syncthreads();

    if (act && lane < 16) {  // ---- softmax, one lane per query row
      float* row = sc + (wid * 16 + lane) * Lp;
      float mx = -1e30f;
      for (int k = 0; k < L; ++k) mx = fmaxf(mx, row[k]);
      float sum = 0.f;
      for (int k = 0; k < L; ++k) { float e = __expf(row[k] - mx); row[k] = e; sum += e; }
      float inv = 1.0f / sum;
      _Float16* pr = pb + (wid * 16 + lane) * Lpp;
      for (int k = 0; k < Lpp; ++k) {  // store P in A-swizzled chunk order
        _Float16 pv = (k < L) ? (_Float16)(row[k] * inv) : (_Float16)0.f;
        pr[(k & ~31) + aswz(k & 31)] = pv;
      }
    }
    __syncthreads();

    if (act) {  // ---- PV: P(16 x Lpp) x V(Lpp x 32)
      v8f o0 = {}, o1 = {};
      const _Float16* pw = pb + (wid * 16 + (lane & 15)) * Lpp;
      for (int kc = 0; kc < nKC; ++kc) {
        AFrag a, b0, b1;
        a.q[0]  = *(const uint4*)&pw[kc * 32 + off16];
        a.q[1]  = *(const uint4*)&pw[kc * 32 + off16 + 8];
        b0.q[0] = *(const uint4*)&vst[ncol * Lpp + kc * 32 + off16];
        b0.q[1] = *(const uint4*)&vst[ncol * Lpp + kc * 32 + off16 + 8];
        b1.q[0] = *(const uint4*)&vst[(16 + ncol) * Lpp + kc * 32 + off16];
        b1.q[1] = *(const uint4*)&vst[(16 + ncol) * Lpp + kc * 32 + off16 + 8];
        o0 = wmma16(a, b0, o0);
        o1 = wmma16(a, b1, o1);
      }
#pragma unroll
      for (int vv = 0; vv < 8; ++vv) {
        int qr = qt * 16 + vv + hi;
        if (qr < L) {
          size_t p = (size_t)(base + qr * STR) * DMODEL + h * DKH;
          O[p + ncol]      = (_Float16)o0[vv];
          O[p + 16 + ncol] = (_Float16)o1[vv];
        }
      }
    }
    __syncthreads();
  }
}

// ---------------------------------------------------------------------------
// Host-side orchestration
// ---------------------------------------------------------------------------
extern "C" void kernel_launch(void* const* d_in, const int* in_sizes, int n_in,
                              void* d_out, int out_size, void* d_ws, size_t ws_size,
                              hipStream_t stream) {
  (void)in_sizes; (void)n_in; (void)out_size; (void)ws_size;
  const float* x   = (const float*)d_in[0];
  const float* spW = (const float*)d_in[1];
  const float* spB = (const float*)d_in[2];
  const float* chW = (const float*)d_in[3];
  const float* chB = (const float*)d_in[4];
  const float* tmW = (const float*)d_in[5];
  const float* tmB = (const float*)d_in[6];
  const float* lng = (const float*)d_in[7];
  const float* lnb = (const float*)d_in[8];
  const float* W1  = (const float*)d_in[9];
  const float* b1  = (const float*)d_in[10];
  const float* W2  = (const float*)d_in[11];
  const float* b2  = (const float*)d_in[12];
  float* out = (float*)d_out;

  // ---- workspace layout (bytes): weights f16 (transposed) | bufA | bufQ/K/V
  char* ws = (char*)d_ws;
  _Float16* hSp = (_Float16*)ws;             // 4 x (256x256), each stored N x K
  _Float16* hCh = hSp + 4 * 65536;
  _Float16* hTm = hCh + 4 * 65536;
  _Float16* hW1 = hTm + 4 * 65536;           // 1024 x 256 (N x K)
  _Float16* hW2 = hW1 + 262144;              // 256 x 1024 (N x K)
  const size_t TOKMAT = (size_t)NTOK * DMODEL;           // 14,155,776 elems
  _Float16* bufA = (_Float16*)(ws + 2621440);            // LN out / attn out
  _Float16* bufQ = bufA + TOKMAT;
  _Float16* bufK = bufQ + TOKMAT;
  _Float16* bufV = bufK + TOKMAT;
  _Float16* bufF = bufQ;   // FFN1 out (55296x1024), aliases Q/K/V region

  // ---- weight conversion + transpose (262144 elems each stack)
  const int WN = 262144;
  dim3 cg((WN + 255) / 256);
  k_cvt_t<<<cg, 256, 0, stream>>>(spW, hSp, DMODEL, DMODEL, 4);
  k_cvt_t<<<cg, 256, 0, stream>>>(chW, hCh, DMODEL, DMODEL, 4);
  k_cvt_t<<<cg, 256, 0, stream>>>(tmW, hTm, DMODEL, DMODEL, 4);
  k_cvt_t<<<cg, 256, 0, stream>>>(W1, hW1, DMODEL, DFF, 1);   // -> 1024 x 256
  k_cvt_t<<<cg, 256, 0, stream>>>(W2, hW2, DFF, DMODEL, 1);   // -> 256 x 1024

  const int M = NTOK;                        // 55296, /128 = 432
  const dim3 gemm256(M / 128, DMODEL / 64);  // (432, 4)
  const dim3 gemmFF1(M / 128, DFF / 64);     // (432, 16)
  const dim3 lnGrid(M / 8);

  const _Float16* Wst[3] = {hSp, hCh, hTm};
  const float*    Bst[3] = {spB, chB, tmB};

  for (int i = 0; i < 3; ++i) {
    const float* src = (i == 0) ? x : out;
    // pre-LN
    k_ln<<<lnGrid, 256, 0, stream>>>(src, lng + 256 * i, lnb + 256 * i, bufA);
    // Q, K, V projections
    k_gemm<0><<<gemm256, 256, 0, stream>>>(bufA, Wst[i] + 0 * 65536, Bst[i] + 0,
                                           (void*)bufQ, nullptr, M, DMODEL, DMODEL);
    k_gemm<0><<<gemm256, 256, 0, stream>>>(bufA, Wst[i] + 1 * 65536, Bst[i] + 256,
                                           (void*)bufK, nullptr, M, DMODEL, DMODEL);
    k_gemm<0><<<gemm256, 256, 0, stream>>>(bufA, Wst[i] + 2 * 65536, Bst[i] + 512,
                                           (void*)bufV, nullptr, M, DMODEL, DMODEL);
    // axial attention -> bufA (f16)
    if (i == 0) {        // LDS: 2*144*64 + 32*160*2 + 4*16*144*4 + 4*16*160*2
      k_attn<0, 4><<<dim3(NB_ * CC_ * TT_, NHEADS), 128, 86016, stream>>>(bufQ, bufK, bufV, bufA);
    } else if (i == 1) { // LDS: 2*16*64 + 32*32*2 + 16*16*4 + 16*32*2
      k_attn<1, 1><<<dim3(NB_ * SS_ * TT_, NHEADS), 32, 6144, stream>>>(bufQ, bufK, bufV, bufA);
    } else {             // LDS: 2*32*64 + 32*32*2 + 2*16*32*4 + 2*16*32*2
      k_attn<2, 2><<<dim3(NB_ * SS_ * CC_, NHEADS), 64, 12288, stream>>>(bufQ, bufK, bufV, bufA);
    }
    // output projection + residual -> out (fp32)
    k_gemm<2><<<gemm256, 256, 0, stream>>>(bufA, Wst[i] + 3 * 65536, Bst[i] + 768,
                                           (void*)out, src, M, DMODEL, DMODEL);
  }

  // ---- FFN: LN -> relu(hW1) -> hW2 + residual
  k_ln<<<lnGrid, 256, 0, stream>>>(out, lng + 768, lnb + 768, bufA);
  k_gemm<1><<<gemmFF1, 256, 0, stream>>>(bufA, hW1, b1, (void*)bufF, nullptr, M, DFF, DMODEL);
  k_gemm<2><<<gemm256, 256, 0, stream>>>(bufF, hW2, b2, (void*)out, out, M, DMODEL, DFF);
}
